// AKT_27917287424232
// MI455X (gfx1250) — compile-verified, matrix operand order checked by reading, not run on hardware
//
#include <hip/hip_runtime.h>

// ---------------- model constants ----------------
#define B_   8
#define S_   1024
#define D_   512
#define H_   8
#define NBS_ (B_ * S_)
#define NEGV (-1.0e32f)
#define SROW 1032            // padded f32 score-row stride in LDS

typedef __bf16 bf16;
typedef __attribute__((ext_vector_type(16))) __bf16 v16bf;
typedef __attribute__((ext_vector_type(8)))  float  v8f;
typedef unsigned int u32;
typedef u32 __attribute__((ext_vector_type(4))) v4u32;
typedef int __attribute__((ext_vector_type(8))) v8i32;
typedef int __attribute__((ext_vector_type(4))) v4i32;

// ---------------- WMMA helpers (CDNA5, wave32) ----------------
static __device__ __forceinline__ v8f wmma_bf16(v16bf a, v16bf b, v8f c) {
  // 8 args: (neg_a, A, neg_b, B, c_mod, C, reuse_a, reuse_b)
  return __builtin_amdgcn_wmma_f32_16x16x32_bf16(false, a, false, b, (short)0, c,
                                                 false, false);
}

// A operand, 16x32 bf16 row-major tile in LDS (ISA 7.12.2): per lane the data is
// two contiguous 8-element runs -> exactly two ds_load_b128.
// Also used for the B operand when the 32x16 tile is staged transposed (N-major).
static __device__ __forceinline__ v16bf frag_ld_a(const bf16* base, int ld) {
  const int lane = threadIdx.x & 31;
  const int m  = lane & 15;
  const int kh = (lane & 16) >> 1;        // lanes>=16 -> K offset 8
  union { v16bf v; uint4 q[2]; } u;
  const bf16* p = base + m * ld + kh;
  u.q[0] = *(const uint4*)p;              // K = kh .. kh+7
  u.q[1] = *(const uint4*)(p + 16);       // K = 16+kh .. 16+kh+7
  return u.v;
}

// A operand sourced from f32 LDS (attention probabilities): 4x ds_load_b128 + cvt.
static __device__ __forceinline__ v16bf frag_ld_a_f32(const float* base, int ld) {
  const int lane = threadIdx.x & 31;
  const int m  = lane & 15;
  const int kh = (lane & 16) >> 1;
  const float* p = base + m * ld + kh;
  const float4 a0 = *(const float4*)p;
  const float4 a1 = *(const float4*)(p + 4);
  const float4 a2 = *(const float4*)(p + 16);
  const float4 a3 = *(const float4*)(p + 20);
  v16bf f;
  f[0]  = (bf16)a0.x; f[1]  = (bf16)a0.y; f[2]  = (bf16)a0.z; f[3]  = (bf16)a0.w;
  f[4]  = (bf16)a1.x; f[5]  = (bf16)a1.y; f[6]  = (bf16)a1.z; f[7]  = (bf16)a1.w;
  f[8]  = (bf16)a2.x; f[9]  = (bf16)a2.y; f[10] = (bf16)a2.z; f[11] = (bf16)a2.w;
  f[12] = (bf16)a3.x; f[13] = (bf16)a3.y; f[14] = (bf16)a3.z; f[15] = (bf16)a3.w;
  return f;
}

// ---------------- TDM: 2D bf16 tile -> LDS (Tensor Data Mover, ISA ch.8) ----------
// amdgpu-toolchain (clang-23) 6-arg form:
//   (uint32x4 g0, int32x8 g1, int32x4 g2, int32x4 g3, int32x8 ext, i32 cpol)
#if __has_builtin(__builtin_amdgcn_tensor_load_to_lds)
static __device__ __forceinline__ void tdm_load_tile_2d_bf16(
    u32 lds_off, const void* gptr, u32 width_elems, u32 rows, u32 row_stride_elems) {
  const unsigned long long ga = (unsigned long long)(uintptr_t)gptr;
  v4u32 g0;
  g0[0] = 1u;                                             // count=1, user D#
  g0[1] = lds_off;                                        // lds_addr (bytes)
  g0[2] = (u32)(ga & 0xffffffffu);                        // global_addr[31:0]
  g0[3] = (u32)((ga >> 32) & 0x01ffffffu) | (2u << 30);   // addr[56:32] | type=2
  v8i32 g1;
  g1[0] = (int)(1u << 16);                 // data_size = 1 (2 bytes/elem)
  g1[1] = (int)(width_elems << 16);        // tensor_dim0[15:0]
  g1[2] = (int)(rows << 16);               // tensor_dim0 hi=0 | tensor_dim1[15:0]
  g1[3] = (int)(width_elems << 16);        // tensor_dim1 hi=0 | tile_dim0
  g1[4] = (int)rows;                       // tile_dim1 (tile_dim2=0)
  g1[5] = (int)row_stride_elems;           // tensor_dim0_stride[31:0]
  g1[6] = 0;
  g1[7] = 0;
  const v4i32 z4 = {0, 0, 0, 0};
  const v8i32 z8 = {0, 0, 0, 0, 0, 0, 0, 0};
  __builtin_amdgcn_tensor_load_to_lds(g0, g1, z4, z4, z8, 0);
}
#endif

// ---------------- small kernels ----------------
__global__ void cvt_kernel(const float* __restrict__ x, bf16* __restrict__ y, int n) {
  int i = blockIdx.x * 256 + threadIdx.x;
  if (i < n) y[i] = (bf16)x[i];
}

static __device__ __forceinline__ float block_sum(float v, float* red) {
  const int lane = threadIdx.x & 31, wave = threadIdx.x >> 5;
#pragma unroll
  for (int d = 16; d > 0; d >>= 1) v += __shfl_xor(v, d, 32);
  if (lane == 0) red[wave] = v;
  __syncthreads();
  float t = red[0] + red[1] + red[2] + red[3];
  __syncthreads();
  return t;
}

// y = LN(res + x) ; writes f32 (into d_out) and bf16 (for next GEMM)
__global__ __launch_bounds__(128)
void ln_res_kernel(const float* __restrict__ res, const float* __restrict__ x,
                   const float* __restrict__ w, const float* __restrict__ b,
                   float* __restrict__ outF, bf16* __restrict__ outB) {
  __shared__ float red[4];
  const int tid = threadIdx.x;
  const size_t base = (size_t)blockIdx.x * D_;
  float v[4];
  float s = 0.f;
#pragma unroll
  for (int q = 0; q < 4; ++q) {
    const int c = tid + q * 128;
    v[q] = res[base + c] + x[base + c];
    s += v[q];
  }
  const float mu = block_sum(s, red) * (1.0f / D_);
  float vs = 0.f;
#pragma unroll
  for (int q = 0; q < 4; ++q) { const float d = v[q] - mu; vs += d * d; }
  const float rstd = rsqrtf(block_sum(vs, red) * (1.0f / D_) + 1e-5f);
#pragma unroll
  for (int q = 0; q < 4; ++q) {
    const int c = tid + q * 128;
    const float y = (v[q] - mu) * rstd * w[c] + b[c];
    outF[base + c] = y;
    outB[base + c] = (bf16)y;
  }
}

// h = hq + o ; hw = h + ow ; cat = [LN(h), LN(hw)] in bf16 (rows of 1024)
__global__ __launch_bounds__(128)
void combine_kernel(const float* __restrict__ hq, const float* __restrict__ o,
                    const float* __restrict__ ow, const float* __restrict__ w,
                    const float* __restrict__ b, bf16* __restrict__ cat) {
  __shared__ float red[4];
  const int tid = threadIdx.x;
  const size_t base = (size_t)blockIdx.x * D_;
  const size_t cb   = (size_t)blockIdx.x * (2 * D_);
  float hv[4], wv[4];
#pragma unroll
  for (int q = 0; q < 4; ++q) {
    const int c = tid + q * 128;
    const float hh = hq[base + c] + o[base + c];
    hv[q] = hh;
    wv[q] = hh + ow[base + c];
  }
  float s = hv[0] + hv[1] + hv[2] + hv[3];
  float mu = block_sum(s, red) * (1.0f / D_);
  float vs = 0.f;
#pragma unroll
  for (int q = 0; q < 4; ++q) { const float d = hv[q] - mu; vs += d * d; }
  float rstd = rsqrtf(block_sum(vs, red) * (1.0f / D_) + 1e-5f);
#pragma unroll
  for (int q = 0; q < 4; ++q) {
    const int c = tid + q * 128;
    cat[cb + c] = (bf16)((hv[q] - mu) * rstd * w[c] + b[c]);
  }
  s = wv[0] + wv[1] + wv[2] + wv[3];
  mu = block_sum(s, red) * (1.0f / D_);
  vs = 0.f;
#pragma unroll
  for (int q = 0; q < 4; ++q) { const float d = wv[q] - mu; vs += d * d; }
  rstd = rsqrtf(block_sum(vs, red) * (1.0f / D_) + 1e-5f);
#pragma unroll
  for (int q = 0; q < 4; ++q) {
    const int c = tid + q * 128;
    cat[cb + D_ + c] = (bf16)((wv[q] - mu) * rstd * w[c] + b[c]);
  }
}

// ---------------- GEMM: Y[M,N] = X[M,K](bf16) @ W[K,N](bf16) + bias ----------------
// block = 128 threads (4 waves), tile 64x64, each wave does 16(M) x 64(N).
// B tile staged transposed (Bst[n][k]) so both operands use 2x ds_load_b128 frags.
__global__ __launch_bounds__(128)
void gemm_kernel(const bf16* __restrict__ X, const bf16* __restrict__ W,
                 const float* __restrict__ bias,
                 float* __restrict__ outF, bf16* __restrict__ outB,
                 int M, int N, int K) {
  __shared__ __align__(16) bf16 As[64 * 32];
  __shared__ __align__(16) bf16 Bst[64 * 32];
  const int tid  = threadIdx.x;
  const int wave = tid >> 5;
  const int lane = tid & 31;
  const int m0 = blockIdx.x * 64;
  const int n0 = blockIdx.y * 64;

  v8f acc[4];
#pragma unroll
  for (int i = 0; i < 4; ++i) acc[i] = (v8f){0.f, 0.f, 0.f, 0.f, 0.f, 0.f, 0.f, 0.f};

  for (int kt = 0; kt < K; kt += 32) {
#pragma unroll
    for (int it = 0; it < 2; ++it) {           // stage A 64x32 row-major
      const int c = tid + it * 128;
      const int r = c >> 2, col = (c & 3) << 3;
      *(uint4*)&As[r * 32 + col] = *(const uint4*)&X[(size_t)(m0 + r) * K + kt + col];
    }
#pragma unroll
    for (int it = 0; it < 2; ++it) {           // stage B 32x64 -> transposed Bst[n][k]
      const int c = tid + it * 128;
      const int r = c >> 3, colb = (c & 7) << 3;
      const uint4 w4 = *(const uint4*)&W[(size_t)(kt + r) * N + n0 + colb];
      const bf16* e = (const bf16*)&w4;
#pragma unroll
      for (int q = 0; q < 8; ++q) Bst[(colb + q) * 32 + r] = e[q];
    }
    __syncthreads();
    const v16bf a = frag_ld_a(&As[wave * 16 * 32], 32);
#pragma unroll
    for (int s = 0; s < 4; ++s) {
      const v16bf b = frag_ld_a(&Bst[(s * 16) * 32], 32);
      acc[s] = wmma_bf16(a, b, acc[s]);
    }
    __syncthreads();
  }
  const int nn = lane & 15;
  const int rext = (lane & 16) >> 1;           // +8 rows for upper half-lanes
#pragma unroll
  for (int s = 0; s < 4; ++s) {
    const int col = n0 + s * 16 + nn;
    const float bv = bias ? bias[col] : 0.f;
#pragma unroll
    for (int r = 0; r < 8; ++r) {
      const int grow = m0 + wave * 16 + r + rext;
      const size_t idx = (size_t)grow * N + col;
      const float v = acc[s][r] + bv;
      if (outF) outF[idx] = v;
      if (outB) outB[idx] = (bf16)v;
    }
  }
}

// ---------------- Attention (distance-decay, kq_same) ----------------
// grid = (S/16 i-tiles, B*H). block = 128 threads (4 waves).
// mode: 0 = j<=i, 1 = j<i, 2 = i-19<=j<=i-1 (window). maxout per reference.
__global__ __launch_bounds__(128)
void attn_kernel(const bf16* __restrict__ QK, const bf16* __restrict__ V,
                 bf16* __restrict__ O, const float* __restrict__ gamma_row,
                 int mode, int maxout) {
  extern __shared__ char smem_raw[];
  float* sS  = (float*)smem_raw;                      // 16 x SROW f32
  bf16*  sQ  = (bf16*)(sS + 16 * SROW);               // 16 x 64 row-major
  bf16*  sK  = sQ + 16 * 64;                          // 4 waves x (16 x 64) row-major
  bf16*  sVt = sK + 4 * 16 * 64;                      // 64(n) x 32(k) transposed

  const int tid  = threadIdx.x;
  const int wave = tid >> 5;
  const int lane = tid & 31;
  const int it0  = blockIdx.x * 16;
  const int bh   = blockIdx.y;
  const int b = bh >> 3, h = bh & (H_ - 1);
  const float g = -fabsf(gamma_row[h]);

  const bf16* Qp = QK + (size_t)b * S_ * D_ + h * 64;
  const bf16* Vp = V  + (size_t)b * S_ * D_ + h * 64;

  // ---- Q tile (16x64) via Tensor Data Mover ----
  const u32 sQ_off = (u32)((const char*)sQ - (const char*)smem_raw) +
                     (u32)__builtin_amdgcn_groupstaticsize();
  (void)sQ_off;
  if (wave == 0) {
#if __has_builtin(__builtin_amdgcn_tensor_load_to_lds)
    tdm_load_tile_2d_bf16(sQ_off, Qp + (size_t)it0 * D_, 64, 16, D_);
#else
#pragma unroll
    for (int q2 = 0; q2 < 4; ++q2) {
      const int c = lane + q2 * 32;
      const int r = c >> 3, col = (c & 7) << 3;
      *(uint4*)&sQ[r * 64 + col] = *(const uint4*)&Qp[(size_t)(it0 + r) * D_ + col];
    }
#endif
  }
  __builtin_amdgcn_s_wait_tensorcnt(0);
  __syncthreads();
  const v16bf aq0 = frag_ld_a(sQ, 64);
  const v16bf aq1 = frag_ld_a(sQ + 32, 64);

  // ---- phase 1: scores S[i, 0..1023] via WMMA, masked, scaled ----
  const int ihi = it0 + 15, ilo = it0;
  for (int t = 0; t < 16; ++t) {
    const int jt = t * 4 + wave;                 // each wave owns distinct j-tiles
    const int j0 = jt * 16;
    bool any;
    if (mode == 0)      any = (j0 <= ihi);
    else if (mode == 1) any = (j0 < ihi);
    else                any = (j0 <= ihi - 1) && (j0 + 15 >= ilo - 19);

    v8f acc = (v8f){0.f, 0.f, 0.f, 0.f, 0.f, 0.f, 0.f, 0.f};
    if (any) {
      bf16* myK = sK + wave * 16 * 64;           // wave-private staging slot
#pragma unroll
      for (int q2 = 0; q2 < 4; ++q2) {
        const int c = lane + q2 * 32;
        const int r = c >> 3, col = (c & 7) << 3;
        *(uint4*)&myK[r * 64 + col] = *(const uint4*)&Qp[(size_t)(j0 + r) * D_ + col];
      }
      asm volatile("s_wait_dscnt 0" ::: "memory");   // wave-local LDS RAW fence
      const v16bf bk0 = frag_ld_a(myK, 64);      // K-tile transposed == A pattern
      const v16bf bk1 = frag_ld_a(myK + 32, 64);
      acc = wmma_bf16(aq0, bk0, acc);
      acc = wmma_bf16(aq1, bk1, acc);
    }
    const int nn = lane & 15;
    const int rext = (lane & 16) >> 1;
#pragma unroll
    for (int r = 0; r < 8; ++r) {
      const int row = r + rext;
      const int i = it0 + row, j = j0 + nn;
      const bool valid = (mode == 0) ? (j <= i)
                       : (mode == 1) ? (j < i)
                                     : (j < i && j >= i - 19);
      sS[row * SROW + j] = valid ? acc[r] * 0.125f : NEGV;
    }
  }
  __syncthreads();

  // ---- phase 2: per-row softmax -> cumsum -> decay -> softmax -> maxout ----
  for (int rr = 0; rr < 4; ++rr) {
    const int row = wave * 4 + rr;
    const int i = it0 + row;
    float* Sr = sS + row * SROW;
    const int c0 = lane * 32;                    // contiguous 32-elem chunk
    float v[32];
#pragma unroll
    for (int c = 0; c < 32; ++c) v[c] = Sr[c0 + c];

    float m1 = -3.0e38f;
#pragma unroll
    for (int c = 0; c < 32; ++c) m1 = fmaxf(m1, v[c]);
#pragma unroll
    for (int d = 16; d > 0; d >>= 1) m1 = fmaxf(m1, __shfl_xor(m1, d, 32));

    float p[32];
    float lsum = 0.f;
#pragma unroll
    for (int c = 0; c < 32; ++c) { p[c] = __expf(v[c] - m1); lsum += p[c]; }
    float tot = lsum;
#pragma unroll
    for (int d = 16; d > 0; d >>= 1) tot += __shfl_xor(tot, d, 32);
    const float invTot = 1.0f / tot;

    float run = lsum;                            // inclusive scan of lane sums
#pragma unroll
    for (int d = 1; d < 32; d <<= 1) {
      const float y = __shfl_up(run, d, 32);
      if (lane >= d) run += y;
    }
    float cum = run - lsum;                      // exclusive prefix
#pragma unroll
    for (int c = 0; c < 32; ++c) {
      cum += p[c];
      const float pos  = fabsf((float)(i - (c0 + c)));
      const float dist = sqrtf(fmaxf(0.f, (1.f - cum * invTot) * pos));
      float wf = __expf(dist * g);
      wf = fminf(fmaxf(wf, 1e-5f), 1e5f);
      v[c] = (v[c] > -1.0e31f) ? v[c] * wf : NEGV;
    }

    float m2 = -3.0e38f;
#pragma unroll
    for (int c = 0; c < 32; ++c) m2 = fmaxf(m2, v[c]);
#pragma unroll
    for (int d = 16; d > 0; d >>= 1) m2 = fmaxf(m2, __shfl_xor(m2, d, 32));
    float s2 = 0.f;
#pragma unroll
    for (int c = 0; c < 32; ++c) { p[c] = __expf(v[c] - m2); s2 += p[c]; }
#pragma unroll
    for (int d = 16; d > 0; d >>= 1) s2 += __shfl_xor(s2, d, 32);
    const float inv2 = 1.0f / s2;
    const float scl  = maxout ? fminf(s2, 5.0f) : 1.0f;  // 1/max(a) == s2
#pragma unroll
    for (int c = 0; c < 32; ++c)
      Sr[c0 + c] = (v[c] > -1.0e31f) ? p[c] * inv2 * scl : 0.f;
  }
  __syncthreads();

  // ---- phase 3: O[16x64] = A[16x1024] @ V[1024x64]; wave -> one 16-col slab ----
  v8f oacc = (v8f){0.f, 0.f, 0.f, 0.f, 0.f, 0.f, 0.f, 0.f};
  for (int kt = 0; kt < 32; ++kt) {
    const int k0 = kt * 32;
#pragma unroll
    for (int q2 = 0; q2 < 2; ++q2) {             // stage V rows k0..k0+31 transposed
      const int c = tid + q2 * 128;
      const int r = c >> 3, colb = (c & 7) << 3;
      const uint4 w4 = *(const uint4*)&Vp[(size_t)(k0 + r) * D_ + colb];
      const bf16* e = (const bf16*)&w4;
#pragma unroll
      for (int q3 = 0; q3 < 8; ++q3) sVt[(colb + q3) * 32 + r] = e[q3];
    }
    __syncthreads();
    const v16bf af = frag_ld_a_f32(sS + k0, SROW);
    const v16bf bv = frag_ld_a(&sVt[(wave * 16) * 32], 32);
    oacc = wmma_bf16(af, bv, oacc);
    __syncthreads();
  }
  const int nn = lane & 15;
  const int rext = (lane & 16) >> 1;
  bf16* Op = O + (size_t)b * S_ * D_ + h * 64 + wave * 16;
#pragma unroll
  for (int r = 0; r < 8; ++r) {
    const int row = r + rext;
    Op[(size_t)(it0 + row) * D_ + nn] = (bf16)oacc[r];
  }
}

// ---------------- host orchestration ----------------
extern "C" void kernel_launch(void* const* d_in, const int* in_sizes, int n_in,
                              void* d_out, int out_size, void* d_ws, size_t ws_size,
                              hipStream_t stream) {
  (void)in_sizes; (void)n_in; (void)out_size; (void)ws_size;
  const float* q_emb = (const float*)d_in[0];
  const float* s_emb = (const float*)d_in[1];
  const float* Wq  = (const float*)d_in[2];
  const float* bq  = (const float*)d_in[3];
  const float* Wqw = (const float*)d_in[4];
  const float* bqw = (const float*)d_in[5];
  const float* Wv  = (const float*)d_in[6];
  const float* bv  = (const float*)d_in[7];
  const float* Wo  = (const float*)d_in[8];
  const float* bo  = (const float*)d_in[9];
  const float* Wow = (const float*)d_in[10];
  const float* bow = (const float*)d_in[11];
  const float* gam = (const float*)d_in[12];
  const float* lnw = (const float*)d_in[13];
  const float* lnb = (const float*)d_in[14];
  const float* Wc  = (const float*)d_in[15];
  const float* bc  = (const float*)d_in[16];

  float* outp = (float*)d_out;                 // [NBS, D]
  float* hqF  = outp + (size_t)NBS_ * D_;      // [NBS, D]
  float* hsF  = hqF  + (size_t)NBS_ * D_;      // [NBS, D]

  char* wsb = (char*)d_ws;
  size_t off = 0;
  auto take = [&](size_t bytes) {
    void* p = wsb + off;
    off += (bytes + 255) & ~(size_t)255;
    return p;
  };
  const size_t DD = (size_t)D_ * D_;
  const size_t AD = (size_t)NBS_ * D_;
  bf16* wq_bf  = (bf16*)take(3 * DD * 2);
  bf16* wqw_bf = (bf16*)take(3 * DD * 2);
  bf16* wv_bf  = (bf16*)take(3 * DD * 2);
  bf16* wo_bf  = (bf16*)take(3 * DD * 2);
  bf16* wow_bf = (bf16*)take(DD * 2);
  bf16* wc_bf  = (bf16*)take(2 * DD * 2);
  bf16* qe_bf  = (bf16*)take(AD * 2);
  bf16* se_bf  = (bf16*)take(AD * 2);
  bf16* t_qk   = (bf16*)take(AD * 2);
  bf16* t_v    = (bf16*)take(AD * 2);
  bf16* t_at   = (bf16*)take(AD * 2);
  bf16* t_qw   = (bf16*)take(AD * 2);
  bf16* t_am   = (bf16*)take(AD * 2);
  bf16* t_aw   = (bf16*)take(AD * 2);
  bf16* hq_bf  = (bf16*)take(AD * 2);
  bf16* hs_bf  = (bf16*)take(AD * 2);
  bf16* cat_bf = (bf16*)take(AD * 2 * 2);
  float* o_f   = (float*)take(AD * 4);
  float* ow_f  = (float*)take(AD * 4);

  auto cvt = [&](const float* s, bf16* dst, size_t n) {
    cvt_kernel<<<dim3((unsigned)((n + 255) / 256)), dim3(256), 0, stream>>>(s, dst, (int)n);
  };
  cvt(q_emb, qe_bf, AD);
  cvt(s_emb, se_bf, AD);
  cvt(Wq,  wq_bf,  3 * DD);
  cvt(Wqw, wqw_bf, 3 * DD);
  cvt(Wv,  wv_bf,  3 * DD);
  cvt(Wo,  wo_bf,  3 * DD);
  cvt(Wow, wow_bf, DD);
  cvt(Wc,  wc_bf,  2 * DD);

  const dim3 gGemm(NBS_ / 64, D_ / 64), bGemm(128);
  auto gemm = [&](const bf16* X, const bf16* W, const float* bias,
                  float* oF, bf16* oB, int K) {
    gemm_kernel<<<gGemm, bGemm, 0, stream>>>(X, W, bias, oF, oB, NBS_, D_, K);
  };

  const size_t attnLds = (size_t)16 * SROW * 4 + 16 * 64 * 2 + 4 * 16 * 64 * 2 + 64 * 32 * 2;
  const dim3 gAttn(S_ / 16, B_ * H_), bAttn(128);
  auto attn = [&](const bf16* QKp, const bf16* Vp, bf16* Op,
                  const float* grow, int mode, int maxo) {
    attn_kernel<<<gAttn, bAttn, attnLds, stream>>>(QKp, Vp, Op, grow, mode, maxo);
  };

  // ---- block 0 (self-attn on q_emb, mask j<=i) ----
  gemm(qe_bf, wq_bf, bq, nullptr, t_qk, D_);
  gemm(qe_bf, wv_bf, bv, nullptr, t_v, D_);
  attn(t_qk, t_v, t_at, gam + 0 * H_, 0, 0);
  gemm(t_at, wo_bf, bo, o_f, nullptr, D_);
  ln_res_kernel<<<NBS_, 128, 0, stream>>>(q_emb, o_f, lnw, lnb, hqF, hq_bf);

  // ---- block 1 (self-attn on s_emb) ----
  gemm(se_bf, wq_bf + DD, bq + D_, nullptr, t_qk, D_);
  gemm(se_bf, wv_bf + DD, bv + D_, nullptr, t_v, D_);
  attn(t_qk, t_v, t_at, gam + 1 * H_, 0, 0);
  gemm(t_at, wo_bf + DD, bo + D_, o_f, nullptr, D_);
  ln_res_kernel<<<NBS_, 128, 0, stream>>>(s_emb, o_f, lnw + D_, lnb + D_, hsF, hs_bf);

  // ---- block 2 (strict-causal maxout + window attention) ----
  gemm(hq_bf, wq_bf + 2 * DD, bq + 2 * D_, nullptr, t_qk, D_);
  gemm(hs_bf, wv_bf + 2 * DD, bv + 2 * D_, nullptr, t_v, D_);
  gemm(hq_bf, wqw_bf + 2 * DD, bqw + 2 * D_, nullptr, t_qw, D_);
  attn(t_qk, t_v, t_am, gam + 2 * H_, 1, 1);
  attn(t_qw, t_v, t_aw, gam + 2 * H_, 2, 0);
  gemm(t_am, wo_bf + 2 * DD, bo + 2 * D_, o_f, nullptr, D_);
  gemm(t_aw, wow_bf, bow, ow_f, nullptr, D_);

  combine_kernel<<<NBS_, 128, 0, stream>>>(hqF, o_f, ow_f, lnw + 2 * D_, lnb + 2 * D_, cat_bf);

  // ---- final: out = cat[NBS,1024] @ Wc[1024,512] + bc ----
  gemm_kernel<<<gGemm, bGemm, 0, stream>>>(cat_bf, wc_bf, bc, outp, nullptr,
                                           NBS_, D_, 2 * D_);
}
